// Attention_22832046145789
// MI455X (gfx1250) — matrix operand order repaired and checked by reading, not compiled, not run on hardware
//
#include <hip/hip_runtime.h>

typedef __bf16 bf16;
typedef __attribute__((ext_vector_type(16))) __bf16 v16bf;
typedef __attribute__((ext_vector_type(8)))  float  v8f;
typedef __attribute__((ext_vector_type(4)))  int    i4v;

#define B_   2
#define T_   2048
#define C_   1024
#define H_   16
#define D_   64
#define ROWS (B_ * T_)     // 4096
#define N3C  (3 * C_)      // 3072

#if __has_builtin(__builtin_amdgcn_global_load_async_to_lds_b128)
#define HAVE_ASYNC_LDS 1
#else
#define HAVE_ASYNC_LDS 0
#endif

// ---------------------------------------------------------------------------
// Global -> LDS 16B copy: async path (ASYNCcnt-tracked, no VGPR round-trip)
// when the builtin exists, otherwise a plain vector copy.
// Builtin signature (from hipcc diagnostic): param0 = AS(1) int4*, 4 args.
// ---------------------------------------------------------------------------
__device__ __forceinline__ void g2lds_b128(bf16* lds_dst, const bf16* gsrc) {
#if HAVE_ASYNC_LDS
  __builtin_amdgcn_global_load_async_to_lds_b128(
      (__attribute__((address_space(1))) i4v*)gsrc,
      (__attribute__((address_space(3))) i4v*)lds_dst, 0, 0);
#else
  *(float4*)lds_dst = *(const float4*)gsrc;
#endif
}

__device__ __forceinline__ void async_fence() {
#if HAVE_ASYNC_LDS
#if __has_builtin(__builtin_amdgcn_s_wait_asynccnt)
  __builtin_amdgcn_s_wait_asynccnt(0);
#else
  asm volatile("s_wait_asynccnt 0x0" ::: "memory");
#endif
#endif
}

// ---------------------------------------------------------------------------
// WMMA helper: D = A(16x32 bf16) * B(32x16 bf16) + C(16x16 f32)
// ---------------------------------------------------------------------------
__device__ __forceinline__ v8f wmma_bf16f32(v16bf a, v16bf b, v8f c) {
  return __builtin_amdgcn_wmma_f32_16x16x32_bf16(
      /*neg_a=*/false, a, /*neg_b=*/false, b,
      /*c_mod=*/(short)0, c, /*reuse_a=*/false, /*reuse_b=*/false);
}

// A fragment (16x32) from row-major [M][K] LDS tile, leading dim ld (elems).
// ISA 16-bit A layout: lane m (m=lane&15) holds row m; lanes<16 hold
// K={0..7,16..23}, lanes>=16 hold K={8..15,24..31}.
__device__ __forceinline__ v16bf frag_a(const bf16* p0, int ld, int m0, int lane) {
  const bf16* p = p0 + (size_t)(m0 + (lane & 15)) * ld + ((lane & 16) ? 8 : 0);
  v16bf a;
#pragma unroll
  for (int e = 0; e < 8; ++e) a[e] = p[e];
#pragma unroll
  for (int e = 0; e < 8; ++e) a[8 + e] = p[16 + e];
  return a;
}

// B fragment (32x16) from LDS tile stored TRANSPOSED as [N][K], ld in elems.
// ISA B layout: lane n (n=lane&15) holds column n; lanes<16 hold
// K={0..7,16..23}, lanes>=16 hold K={8..15,24..31} (same order as A).
__device__ __forceinline__ v16bf frag_b(const bf16* p0, int ld, int n0, int lane) {
  const bf16* p = p0 + (size_t)(n0 + (lane & 15)) * ld + ((lane & 16) ? 8 : 0);
  v16bf b;
#pragma unroll
  for (int e = 0; e < 8; ++e) b[e] = p[e];
#pragma unroll
  for (int e = 0; e < 8; ++e) b[8 + e] = p[16 + e];
  return b;
}

// Pack two bf16 into one dword for b32 LDS stores.
union pk2 { bf16 h[2]; unsigned u; };

// ---------------------------------------------------------------------------
// Kernel 1: LayerNorm (eps=1e-6) + cast to bf16.  One block per row of 1024.
// ---------------------------------------------------------------------------
__global__ void __launch_bounds__(256) k_layernorm(
    const float* __restrict__ x, const float* __restrict__ gamma,
    const float* __restrict__ beta, bf16* __restrict__ xn) {
  __shared__ float red[256];
  const int row = blockIdx.x;
  const int tid = threadIdx.x;
  const float4 v = ((const float4*)(x + (size_t)row * C_))[tid];
  float s = v.x + v.y + v.z + v.w;
  red[tid] = s;
  __syncthreads();
#pragma unroll
  for (int off = 128; off > 0; off >>= 1) {
    if (tid < off) red[tid] += red[tid + off];
    __syncthreads();
  }
  const float mu = red[0] * (1.0f / (float)C_);
  __syncthreads();
  const float d0 = v.x - mu, d1 = v.y - mu, d2 = v.z - mu, d3 = v.w - mu;
  red[tid] = d0 * d0 + d1 * d1 + d2 * d2 + d3 * d3;
  __syncthreads();
#pragma unroll
  for (int off = 128; off > 0; off >>= 1) {
    if (tid < off) red[tid] += red[tid + off];
    __syncthreads();
  }
  const float rstd = rsqrtf(red[0] * (1.0f / (float)C_) + 1e-6f);
  const int c = tid * 4;
  bf16* o = xn + (size_t)row * C_ + c;
  o[0] = (bf16)(d0 * rstd * gamma[c + 0] + beta[c + 0]);
  o[1] = (bf16)(d1 * rstd * gamma[c + 1] + beta[c + 1]);
  o[2] = (bf16)(d2 * rstd * gamma[c + 2] + beta[c + 2]);
  o[3] = (bf16)(d3 * rstd * gamma[c + 3] + beta[c + 3]);
}

// ---------------------------------------------------------------------------
// Kernel 2: f32 -> bf16 cast (weights)
// ---------------------------------------------------------------------------
__global__ void k_cast_bf16(const float* __restrict__ in, bf16* __restrict__ out, int n) {
  int i = blockIdx.x * blockDim.x + threadIdx.x;
  const int stride = gridDim.x * blockDim.x;
  for (; i < n; i += stride) out[i] = (bf16)in[i];
}

// ---------------------------------------------------------------------------
// Kernel 3: QKV GEMM.  [4096 x 1024]bf16 @ [1024 x 3072]bf16 + bias.
// Block tile 64(M) x 128(N), 8 waves of 32x32, K step 32.
// A tile staged via async global->LDS; B tile transposed with packed b32
// stores (two K-rows per thread).
// ---------------------------------------------------------------------------
#define LDW 40  // LDS leading dim (32 + 8 pad) -> 80B rows, 16B aligned
__global__ void __launch_bounds__(256) k_gemm_qkv(
    const bf16* __restrict__ A, const bf16* __restrict__ W,
    const float* __restrict__ bias,
    bf16* __restrict__ Qo, bf16* __restrict__ Ko, bf16* __restrict__ Vo) {
  __shared__ __align__(16) bf16 sA[64 * LDW];    // [m][k]
  __shared__ __align__(16) bf16 sB[128 * LDW];   // [n][k] (transposed)
  const int tid = threadIdx.x, lane = tid & 31, wave = tid >> 5;
  const int wm = wave & 1, wn = wave >> 1;       // 2 x 4 wave grid
  const int m0 = blockIdx.x * 64;
  const int n0 = blockIdx.y * 128;
  v8f acc[2][2] = {};
  for (int k0 = 0; k0 < C_; k0 += 32) {
    __syncthreads();
    {  // A tile 64x32: 256 threads x one 16B chunk, async to LDS
      const int r = tid >> 2, ks = (tid & 3) * 8;
      g2lds_b128(sA + r * LDW + ks, A + (size_t)(m0 + r) * C_ + k0 + ks);
    }
    {  // B tile 32x128: two K-rows per thread, packed b32 transposed stores
      const int kk = (tid >> 4) * 2, ns = (tid & 15) * 8;
      const bf16* g0 = W + (size_t)(k0 + kk) * N3C + n0 + ns;
      const bf16* g1 = g0 + N3C;
      bf16 r0[8], r1[8];
      *(float4*)r0 = *(const float4*)g0;
      *(float4*)r1 = *(const float4*)g1;
#pragma unroll
      for (int i = 0; i < 8; ++i) {
        pk2 p; p.h[0] = r0[i]; p.h[1] = r1[i];
        *(unsigned*)(sB + (ns + i) * LDW + kk) = p.u;
      }
    }
    if (k0 + 32 < C_) {
      __builtin_prefetch(A + (size_t)(m0 + (tid >> 2)) * C_ + k0 + 32, 0, 1);
      __builtin_prefetch(W + (size_t)(k0 + 32 + (tid >> 4) * 2) * N3C + n0, 0, 1);
    }
    async_fence();
    __syncthreads();
    const v16bf a0 = frag_a(sA, LDW, wm * 32, lane);
    const v16bf a1 = frag_a(sA, LDW, wm * 32 + 16, lane);
    const v16bf b0 = frag_b(sB, LDW, wn * 32, lane);
    const v16bf b1 = frag_b(sB, LDW, wn * 32 + 16, lane);
    acc[0][0] = wmma_bf16f32(a0, b0, acc[0][0]);
    acc[0][1] = wmma_bf16f32(a0, b1, acc[0][1]);
    acc[1][0] = wmma_bf16f32(a1, b0, acc[1][0]);
    acc[1][1] = wmma_bf16f32(a1, b1, acc[1][1]);
  }
#pragma unroll
  for (int tm = 0; tm < 2; ++tm)
#pragma unroll
    for (int tn = 0; tn < 2; ++tn) {
      const int ng = n0 + wn * 32 + tn * 16 + (lane & 15);
      const float bia = bias[ng];
      const int sec = ng >> 10;          // 0=Q 1=K 2=V
      const int c2 = ng & (C_ - 1);
      const int h = c2 >> 6, d = c2 & 63;
      bf16* dst = (sec == 0) ? Qo : ((sec == 1) ? Ko : Vo);
      const float scale = (sec == 0) ? 0.125f : 1.0f;  // 1/(TEMP*sqrt(64))
      const int mg = m0 + wm * 32 + tm * 16 + ((lane & 16) ? 8 : 0);
#pragma unroll
      for (int r = 0; r < 8; ++r) {
        const int m = mg + r;
        const int b = m >> 11, t = m & (T_ - 1);
        dst[(((size_t)b * H_ + h) * T_ + t) * D_ + d] =
            (bf16)((acc[tm][tn][r] + bia) * scale);
      }
    }
}

// ---------------------------------------------------------------------------
// Kernel 4: flash-style causal attention. Block = 4 waves = 64 Q rows of one
// (b,h). Wave owns 16 Q rows x 64 d. K tiles of 64 keys streamed via LDS
// (async), V transposed with packed b32 stores.
// ---------------------------------------------------------------------------
#define LDT 80  // key/V/P LDS leading dim (64 + 16 pad) -> 160B rows
__global__ void __launch_bounds__(128) k_attention(
    const bf16* __restrict__ Q, const bf16* __restrict__ K,
    const bf16* __restrict__ V, bf16* __restrict__ AO) {
  __shared__ __align__(16) bf16 sK[64 * LDT];          // [key][d]
  __shared__ __align__(16) bf16 sV[64 * LDT];          // [d][key]  (transposed)
  __shared__ __align__(16) bf16 sP[4 * 16 * LDT];      // per-wave P staging
  const int tid = threadIdx.x, lane = tid & 31, wave = tid >> 5;
  const int nqt = T_ / 64;                              // 32
  const int qt = blockIdx.x % nqt;
  const int h  = (blockIdx.x / nqt) % H_;
  const int b  = blockIdx.x / (nqt * H_);
  const size_t bh = ((size_t)b * H_ + h) * T_ * D_;
  const bf16* Qbh = Q + bh;
  const bf16* Kbh = K + bh;
  const bf16* Vbh = V + bh;
  const int q0 = qt * 64;

  // Q fragments for this wave, loaded once, directly in A layout (d = 0..63).
  v16bf qf0, qf1;
  {
    const bf16* p = Qbh + (size_t)(q0 + wave * 16 + (lane & 15)) * D_ +
                    ((lane & 16) ? 8 : 0);
#pragma unroll
    for (int e = 0; e < 8; ++e) {
      qf0[e] = p[e];       qf0[8 + e] = p[16 + e];
      qf1[e] = p[32 + e];  qf1[8 + e] = p[48 + e];
    }
  }

  float mrun[8], lrun[8];
  v8f oacc[4] = {};
#pragma unroll
  for (int r = 0; r < 8; ++r) { mrun[r] = -1e30f; lrun[r] = 0.0f; }

  for (int kt = 0; kt <= qt; ++kt) {
    __syncthreads();
    {  // K tile [key][d]: async copy, 128 thr x half row (4 x 16B)
      const int kr = tid >> 1, dss = (tid & 1) * 32;
      const bf16* gk = Kbh + (size_t)(kt * 64 + kr) * D_ + dss;
      bf16* dk = sK + kr * LDT + dss;
      g2lds_b128(dk,      gk);
      g2lds_b128(dk + 8,  gk + 8);
      g2lds_b128(dk + 16, gk + 16);
      g2lds_b128(dk + 24, gk + 24);
    }
    {  // V tile transposed [d][key]: two keys per thread, packed b32 stores
      const int kk = (tid >> 2) * 2, ds = (tid & 3) * 16;
      const bf16* gv0 = Vbh + (size_t)(kt * 64 + kk) * D_ + ds;
      const bf16* gv1 = gv0 + D_;
      bf16 r0[16], r1[16];
      ((float4*)r0)[0] = ((const float4*)gv0)[0];
      ((float4*)r0)[1] = ((const float4*)gv0)[1];
      ((float4*)r1)[0] = ((const float4*)gv1)[0];
      ((float4*)r1)[1] = ((const float4*)gv1)[1];
#pragma unroll
      for (int i = 0; i < 16; ++i) {
        pk2 p; p.h[0] = r0[i]; p.h[1] = r1[i];
        *(unsigned*)(sV + (ds + i) * LDT + kk) = p.u;
      }
    }
    async_fence();
    __syncthreads();

    // S = Q @ K^T : 4 tiles of 16x16, reduction over d = 64 (2 WMMAs each)
    v8f s[4];
#pragma unroll
    for (int j = 0; j < 4; ++j) {
      v8f a = {};
      a = wmma_bf16f32(qf0, frag_b(sK, LDT, j * 16, lane), a);       // d 0..31
      a = wmma_bf16f32(qf1, frag_b(sK + 32, LDT, j * 16, lane), a);  // d 32..63
      s[j] = a;
    }

    if (kt == qt) {  // causal mask on the diagonal tile
#pragma unroll
      for (int j = 0; j < 4; ++j) {
        const int colg = kt * 64 + j * 16 + (lane & 15);
        const int rg0 = q0 + wave * 16 + ((lane & 16) ? 8 : 0);
#pragma unroll
        for (int r = 0; r < 8; ++r)
          if (colg > rg0 + r) s[j][r] = -1e30f;
      }
    }

    // Online softmax: row state lives per 16-lane half (rows 0..7 / 8..15).
    float mnew[8], alpha[8];
#pragma unroll
    for (int r = 0; r < 8; ++r) {
      float v = fmaxf(fmaxf(s[0][r], s[1][r]), fmaxf(s[2][r], s[3][r]));
#pragma unroll
      for (int off = 1; off < 16; off <<= 1) v = fmaxf(v, __shfl_xor(v, off, 32));
      mnew[r] = fmaxf(mrun[r], v);
      alpha[r] = __expf(mrun[r] - mnew[r]);
      mrun[r] = mnew[r];
    }

    bf16* pw = sP + wave * 16 * LDT;
#pragma unroll
    for (int r = 0; r < 8; ++r) {
      const int row = ((lane & 16) ? 8 : 0) + r;
      float ssum = 0.0f;
#pragma unroll
      for (int j = 0; j < 4; ++j) {
        const float p = __expf(s[j][r] - mnew[r]);
        pw[row * LDT + j * 16 + (lane & 15)] = (bf16)p;
        ssum += p;
      }
#pragma unroll
      for (int off = 1; off < 16; off <<= 1) ssum += __shfl_xor(ssum, off, 32);
      lrun[r] = lrun[r] * alpha[r] + ssum;
#pragma unroll
      for (int j = 0; j < 4; ++j) oacc[j][r] *= alpha[r];
    }

    // per-wave LDS RAW fence before re-reading P in A layout
    asm volatile("s_wait_dscnt 0x0" ::: "memory");

    const v16bf pa0 = frag_a(pw, LDT, 0, lane);        // keys 0..31
    const v16bf pa1 = frag_a(pw + 32, LDT, 0, lane);   // keys 32..63
#pragma unroll
    for (int j = 0; j < 4; ++j) {
      oacc[j] = wmma_bf16f32(pa0, frag_b(sV, LDT, j * 16, lane), oacc[j]);
      oacc[j] = wmma_bf16f32(pa1, frag_b(sV + 32, LDT, j * 16, lane), oacc[j]);
    }
  }

  // normalize and emit [B,T,C] bf16
  const int rb = (lane & 16) ? 8 : 0;
#pragma unroll
  for (int j = 0; j < 4; ++j)
#pragma unroll
    for (int r = 0; r < 8; ++r) {
      const int t = q0 + wave * 16 + rb + r;
      const int c = h * 64 + j * 16 + (lane & 15);
      AO[((size_t)b * T_ + t) * C_ + c] = (bf16)(oacc[j][r] / lrun[r]);
    }
}

// ---------------------------------------------------------------------------
// Kernel 5: output projection GEMM, f32 result + bias.
// ---------------------------------------------------------------------------
__global__ void __launch_bounds__(256) k_gemm_proj(
    const bf16* __restrict__ A, const bf16* __restrict__ W,
    const float* __restrict__ bias, float* __restrict__ O) {
  __shared__ __align__(16) bf16 sA[64 * LDW];
  __shared__ __align__(16) bf16 sB[128 * LDW];
  const int tid = threadIdx.x, lane = tid & 31, wave = tid >> 5;
  const int wm = wave & 1, wn = wave >> 1;
  const int m0 = blockIdx.x * 64;
  const int n0 = blockIdx.y * 128;
  v8f acc[2][2] = {};
  for (int k0 = 0; k0 < C_; k0 += 32) {
    __syncthreads();
    {
      const int r = tid >> 2, ks = (tid & 3) * 8;
      g2lds_b128(sA + r * LDW + ks, A + (size_t)(m0 + r) * C_ + k0 + ks);
    }
    {
      const int kk = (tid >> 4) * 2, ns = (tid & 15) * 8;
      const bf16* g0 = W + (size_t)(k0 + kk) * C_ + n0 + ns;
      const bf16* g1 = g0 + C_;
      bf16 r0[8], r1[8];
      *(float4*)r0 = *(const float4*)g0;
      *(float4*)r1 = *(const float4*)g1;
#pragma unroll
      for (int i = 0; i < 8; ++i) {
        pk2 p; p.h[0] = r0[i]; p.h[1] = r1[i];
        *(unsigned*)(sB + (ns + i) * LDW + kk) = p.u;
      }
    }
    if (k0 + 32 < C_) {
      __builtin_prefetch(A + (size_t)(m0 + (tid >> 2)) * C_ + k0 + 32, 0, 1);
      __builtin_prefetch(W + (size_t)(k0 + 32 + (tid >> 4) * 2) * C_ + n0, 0, 1);
    }
    async_fence();
    __syncthreads();
    const v16bf a0 = frag_a(sA, LDW, wm * 32, lane);
    const v16bf a1 = frag_a(sA, LDW, wm * 32 + 16, lane);
    const v16bf b0 = frag_b(sB, LDW, wn * 32, lane);
    const v16bf b1 = frag_b(sB, LDW, wn * 32 + 16, lane);
    acc[0][0] = wmma_bf16f32(a0, b0, acc[0][0]);
    acc[0][1] = wmma_bf16f32(a0, b1, acc[0][1]);
    acc[1][0] = wmma_bf16f32(a1, b0, acc[1][0]);
    acc[1][1] = wmma_bf16f32(a1, b1, acc[1][1]);
  }
#pragma unroll
  for (int tm = 0; tm < 2; ++tm)
#pragma unroll
    for (int tn = 0; tn < 2; ++tn) {
      const int ng = n0 + wn * 32 + tn * 16 + (lane & 15);
      const float bia = bias[ng];
      const int mg = m0 + wm * 32 + tm * 16 + ((lane & 16) ? 8 : 0);
#pragma unroll
      for (int r = 0; r < 8; ++r)
        O[(size_t)(mg + r) * C_ + ng] = acc[tm][tn][r] + bia;
    }
}

// ---------------------------------------------------------------------------
// Host launcher
// ---------------------------------------------------------------------------
extern "C" void kernel_launch(void* const* d_in, const int* in_sizes, int n_in,
                              void* d_out, int out_size, void* d_ws, size_t ws_size,
                              hipStream_t stream) {
  (void)in_sizes; (void)n_in; (void)out_size; (void)ws_size;
  const float* x        = (const float*)d_in[0];
  /* d_in[1] = causal mask: implemented analytically, not read */
  const float* ln_scale = (const float*)d_in[2];
  const float* ln_bias  = (const float*)d_in[3];
  const float* qkv_w    = (const float*)d_in[4];
  const float* qkv_b    = (const float*)d_in[5];
  const float* proj_w   = (const float*)d_in[6];
  const float* proj_b   = (const float*)d_in[7];
  float* out = (float*)d_out;

  char* ws = (char*)d_ws;
  size_t off = 0;
  auto carve = [&](size_t bytes) -> char* {
    char* p = ws + off;
    off += (bytes + 255) & ~(size_t)255;
    return p;
  };
  bf16* xn = (bf16*)carve((size_t)ROWS * C_ * sizeof(bf16));      // 8 MB
  bf16* wq = (bf16*)carve((size_t)C_ * N3C * sizeof(bf16));       // 6 MB
  bf16* wp = (bf16*)carve((size_t)C_ * C_ * sizeof(bf16));        // 2 MB
  bf16* Qb = (bf16*)carve((size_t)ROWS * C_ * sizeof(bf16));      // 8 MB
  bf16* Kb = (bf16*)carve((size_t)ROWS * C_ * sizeof(bf16));      // 8 MB
  bf16* Vb = (bf16*)carve((size_t)ROWS * C_ * sizeof(bf16));      // 8 MB
  bf16* ao = (bf16*)carve((size_t)ROWS * C_ * sizeof(bf16));      // 8 MB

  k_layernorm<<<ROWS, 256, 0, stream>>>(x, ln_scale, ln_bias, xn);
  k_cast_bf16<<<2048, 256, 0, stream>>>(qkv_w, wq, C_ * N3C);
  k_cast_bf16<<<1024, 256, 0, stream>>>(proj_w, wp, C_ * C_);
  k_gemm_qkv<<<dim3(ROWS / 64, N3C / 128), 256, 0, stream>>>(xn, wq, qkv_b,
                                                             Qb, Kb, Vb);
  k_attention<<<B_ * H_ * (T_ / 64), 128, 0, stream>>>(Qb, Kb, Vb, ao);
  k_gemm_proj<<<dim3(ROWS / 64, C_ / 128), 256, 0, stream>>>(ao, wp, proj_b, out);
}
